// CosineSimCodebook_47768626266325
// MI455X (gfx1250) — matrix operand order repaired and checked by reading, not compiled, not run on hardware
//
#include <hip/hip_runtime.h>
#include <hip/hip_bf16.h>

// ---------------------------------------------------------------------------
// CDNA5 (gfx1250) vector-quantization: L2-normalize -> bf16 WMMA GEMM-argmax
// -> gather.  Wave32, v_wmma_f32_16x16x32_bf16, async global->LDS staging.
// ---------------------------------------------------------------------------

typedef __attribute__((ext_vector_type(16))) __bf16 v16bf;
typedef __attribute__((ext_vector_type(8)))  float  v8f;
typedef __attribute__((ext_vector_type(4)))  int    v4i;

#define D_DIM   256      // feature dim (fixed by reference)
#define MT      128      // query rows per workgroup tile
#define CT      128      // codebook rows per LDS chunk
#define LDA     264      // padded LDS row stride in bf16 elems (528B, 16B-mult)
#define CSPLIT  4        // C-dimension split across workgroups

#ifdef __has_builtin
#if __has_builtin(__builtin_amdgcn_global_load_async_to_lds_b128)
#define USE_ASYNC_LDS 1
#endif
#endif
#ifndef USE_ASYNC_LDS
#define USE_ASYNC_LDS 0
#endif

__device__ __forceinline__ unsigned f2key(float f) {
  unsigned u = __float_as_uint(f);
  return (u & 0x80000000u) ? ~u : (u | 0x80000000u);  // monotonic float->uint
}

// ---------------------------------------------------------------------------
// Kernel 1: per-row L2 normalize fp32 -> bf16.  One wave per row.
// ---------------------------------------------------------------------------
__global__ __launch_bounds__(256) void nrm_bf16_kernel(
    const float* __restrict__ src, __bf16* __restrict__ dst, int rows) {
  const int wave = threadIdx.x >> 5;
  const int lane = threadIdx.x & 31;
  const int row  = blockIdx.x * 8 + wave;
  if (row >= rows) return;

  const float* p = src + (size_t)row * D_DIM + lane * 8;
  float4 a = *(const float4*)p;
  float4 b = *(const float4*)(p + 4);
  float ss = a.x * a.x + a.y * a.y + a.z * a.z + a.w * a.w +
             b.x * b.x + b.y * b.y + b.z * b.z + b.w * b.w;
#pragma unroll
  for (int off = 16; off >= 1; off >>= 1) ss += __shfl_xor(ss, off, 32);
  const float inv = 1.0f / fmaxf(sqrtf(ss), 1e-12f);

  union { __bf16 h[8]; float4 v; } u;
  u.h[0] = (__bf16)(a.x * inv); u.h[1] = (__bf16)(a.y * inv);
  u.h[2] = (__bf16)(a.z * inv); u.h[3] = (__bf16)(a.w * inv);
  u.h[4] = (__bf16)(b.x * inv); u.h[5] = (__bf16)(b.y * inv);
  u.h[6] = (__bf16)(b.z * inv); u.h[7] = (__bf16)(b.w * inv);
  *(float4*)(dst + (size_t)row * D_DIM + lane * 8) = u.v;
}

// ---------------------------------------------------------------------------
// Kernel 2: init best[] (packed (key(score)<<32)|~index) to 0.
// ---------------------------------------------------------------------------
__global__ void init_best_kernel(unsigned long long* best, int M) {
  int i = blockIdx.x * blockDim.x + threadIdx.x;
  if (i < M) best[i] = 0ull;
}

// ---------------------------------------------------------------------------
// Cooperative tile copy: ROWS x 256 bf16, global -> LDS (padded stride LDA).
// Uses async-to-LDS (ASYNCcnt) when available.
// ---------------------------------------------------------------------------
template <int ROWS>
__device__ __forceinline__ void issue_tile(__bf16* __restrict__ ldst,
                                           const __bf16* __restrict__ gsrc,
                                           int tid) {
#pragma unroll
  for (int i = 0; i < ROWS / 8; ++i) {
    const int q   = tid + i * 256;   // float4 index; coalesced across tid
    const int row = q >> 5;          // 32 float4 per 256-elem row
    const int col = (q & 31) * 8;    // bf16 column
    const __bf16* g = gsrc + (size_t)row * D_DIM + col;
    __bf16*       l = ldst + row * LDA + col;
#if USE_ASYNC_LDS
    __builtin_amdgcn_global_load_async_to_lds_b128(
        (__attribute__((address_space(1))) v4i*)g,
        (__attribute__((address_space(3))) v4i*)l, 0, 0);
#else
    *(float4*)l = *(const float4*)g;
#endif
  }
}

// ---------------------------------------------------------------------------
// Kernel 3: GEMM-argmax.  Grid = (M/128, CSPLIT).  8 waves/WG.
// Wave grid 4(M) x 2(C): wave tile = 32 rows x 64 cols = 2x4 WMMA tiles.
// ---------------------------------------------------------------------------
__global__ __launch_bounds__(256, 1) void vq_argmax_kernel(
    const __bf16* __restrict__ Xn, const __bf16* __restrict__ En,
    unsigned long long* __restrict__ best, int M, int C) {
  __shared__ __align__(16) __bf16 lA[MT * LDA];        // 67.6 KB
  __shared__ __align__(16) __bf16 lB[2][CT * LDA];     // 2 x 67.6 KB
  __shared__ unsigned long long  lBest[MT];            // 1 KB

  const int tid  = threadIdx.x;
  const int lane = tid & 31;
  const int wave = tid >> 5;
  const int wm   = wave & 3;    // 0..3  -> row block of 32
  const int wc   = wave >> 2;   // 0..1  -> col block of 64
  const int hi   = lane >> 4;   // half-wave select
  const int l15  = lane & 15;

  const int mBase   = blockIdx.x * MT;
  const int cPer    = C / CSPLIT;
  const int cBeg    = blockIdx.y * cPer;
  const int nChunks = cPer / CT;

  if (tid < MT) lBest[tid] = 0ull;

  issue_tile<MT>(lA, Xn + (size_t)mBase * D_DIM, tid);      // A tile, once
  issue_tile<CT>(lB[0], En + (size_t)cBeg * D_DIM, tid);    // first B chunk
#if USE_ASYNC_LDS
  asm volatile("s_wait_asynccnt 0" ::: "memory");
#endif
  __syncthreads();

  const v8f vzero = {0.f, 0.f, 0.f, 0.f, 0.f, 0.f, 0.f, 0.f};

  for (int ch = 0; ch < nChunks; ++ch) {
    const __bf16* Bt = lB[ch & 1];
    const bool more = (ch + 1 < nChunks);
    if (more)  // prefetch next chunk while computing this one
      issue_tile<CT>(lB[(ch + 1) & 1],
                     En + (size_t)(cBeg + (ch + 1) * CT) * D_DIM, tid);

    v8f acc[2][4];
#pragma unroll
    for (int tm = 0; tm < 2; ++tm)
#pragma unroll
      for (int tc = 0; tc < 4; ++tc) acc[tm][tc] = vzero;

    // K loop: D=256 in 8 steps of 32 (bf16 WMMA K)
#pragma unroll
    for (int k = 0; k < D_DIM; k += 32) {
      v16bf afrag[2];
#pragma unroll
      for (int tm = 0; tm < 2; ++tm) {
        // A 16x32 bf16 layout: lane m (0-15): K=0..7,16..23; lane m+16: +8
        const __bf16* pa = lA + (wm * 32 + tm * 16 + l15) * LDA + k + hi * 8;
        float4* va = (float4*)&afrag[tm];
        va[0] = *(const float4*)pa;          // K +0..7
        va[1] = *(const float4*)(pa + 16);   // K +16..23
      }
#pragma unroll
      for (int tc = 0; tc < 4; ++tc) {
        // B 32x16 bf16 layout: lane n (0-15): K=0..7,16..23; lane n+16: +8
        const __bf16* pb = Bt + (wc * 64 + tc * 16 + l15) * LDA + k + hi * 8;
        v16bf bfrag;
        float4* vb = (float4*)&bfrag;
        vb[0] = *(const float4*)pb;
        vb[1] = *(const float4*)(pb + 16);
        acc[0][tc] = __builtin_amdgcn_wmma_f32_16x16x32_bf16(
            false, afrag[0], false, bfrag, (short)0, acc[0][tc], false, false);
        acc[1][tc] = __builtin_amdgcn_wmma_f32_16x16x32_bf16(
            false, afrag[1], false, bfrag, (short)0, acc[1][tc], false, false);
      }
    }

    // Per-chunk argmax: C/D tile layout => VGPR r, lanes 0-15: row r, col l;
    // lanes 16-31: row r+8, col l-16.
    const int cChunk = cBeg + ch * CT + wc * 64;
#pragma unroll
    for (int tm = 0; tm < 2; ++tm) {
#pragma unroll
      for (int r = 0; r < 8; ++r) {
        float v  = acc[tm][0][r];
        int   ci = cChunk + l15;
#pragma unroll
        for (int tc = 1; tc < 4; ++tc) {
          float x  = acc[tm][tc][r];
          int   c2 = cChunk + tc * 16 + l15;
          if (x > v || (x == v && c2 < ci)) { v = x; ci = c2; }
        }
#pragma unroll
        for (int off = 8; off >= 1; off >>= 1) {   // reduce within 16-lane half
          float ov = __shfl_xor(v, off, 32);
          int   oi = __shfl_xor(ci, off, 32);
          if (ov > v || (ov == v && oi < ci)) { v = ov; ci = oi; }
        }
        if (l15 == 0) {
          const int rowL = wm * 32 + tm * 16 + r + hi * 8;
          const unsigned long long pk =
              ((unsigned long long)f2key(v) << 32) | (unsigned)(~ci);
          atomicMax(&lBest[rowL], pk);   // ds_max_u64
        }
      }
    }

#if USE_ASYNC_LDS
    if (more) asm volatile("s_wait_asynccnt 0" ::: "memory");
#endif
    __syncthreads();
  }

  if (tid < MT) atomicMax(&best[mBase + tid], lBest[tid]);  // merge C-splits
}

// ---------------------------------------------------------------------------
// Kernel 4: gather winning (unnormalized) codebook row + emit index.
// ---------------------------------------------------------------------------
__global__ __launch_bounds__(64) void gather_kernel(
    const unsigned long long* __restrict__ best,
    const float* __restrict__ embed, float* __restrict__ quant,
    float* __restrict__ indOut, int M) {
  const int m = blockIdx.x;
  const unsigned long long b = best[m];
  const int idx = (int)(~(unsigned)b) & 0x7FFFFFFF;  // undo ~index packing
  const float4* src = (const float4*)(embed + (size_t)idx * D_DIM);
  float4*       dst = (float4*)(quant + (size_t)m * D_DIM);
  for (int t = threadIdx.x; t < D_DIM / 4; t += blockDim.x) dst[t] = src[t];
  if (threadIdx.x == 0) indOut[m] = (float)idx;
}

// ---------------------------------------------------------------------------
extern "C" void kernel_launch(void* const* d_in, const int* in_sizes, int n_in,
                              void* d_out, int out_size, void* d_ws,
                              size_t ws_size, hipStream_t stream) {
  const float* x     = (const float*)d_in[0];  // [B,N,D] fp32
  const float* embed = (const float*)d_in[1];  // [1,C,D] fp32
  const int M = in_sizes[0] / D_DIM;           // 16384
  const int C = in_sizes[1] / D_DIM;           // 16384

  char* ws = (char*)d_ws;
  __bf16* Xn = (__bf16*)ws;                                   // 8 MB
  __bf16* En = (__bf16*)(ws + (size_t)M * D_DIM * 2);         // 8 MB
  unsigned long long* best =
      (unsigned long long*)(ws + (size_t)(M + C) * D_DIM * 2);  // 128 KB

  nrm_bf16_kernel<<<M / 8, 256, 0, stream>>>(x, Xn, M);
  nrm_bf16_kernel<<<C / 8, 256, 0, stream>>>(embed, En, C);
  init_best_kernel<<<(M + 255) / 256, 256, 0, stream>>>(best, M);

  dim3 grid(M / MT, CSPLIT);
  vq_argmax_kernel<<<grid, 256, 0, stream>>>(Xn, En, best, M, C);

  float* quant  = (float*)d_out;
  float* indOut = quant + (size_t)M * D_DIM;
  gather_kernel<<<M, 64, 0, stream>>>(best, embed, quant, indOut, M);
}
